// TreeCell_56762287784428
// MI455X (gfx1250) — matrix operand order repaired and checked by reading, not compiled, not run on hardware
//
#include <hip/hip_runtime.h>

// -------- CDNA5 WMMA vector types --------
typedef __attribute__((ext_vector_type(16))) __bf16 v16bf;
typedef __attribute__((ext_vector_type(8)))  float  v8f;

#define B_SZ   65536
#define H_SZ   256
#define K_TOT  768            // I + C*H = 256*3
#define KB_N   24             // K_TOT / 32
#define NT_N   16             // H / 16
#define G_N    5
#define BH     ((long)B_SZ * H_SZ)
#define M_TILE 32
#define XSTRIDE 784           // 768 + 16 pad (bf16 elems), 1568B row stride (16B aligned)

// round-to-nearest-even f32 -> bf16
__device__ __forceinline__ unsigned f2bf(float f) {
  unsigned u = __float_as_uint(f);
  unsigned r = u + 0x7FFFu + ((u >> 16) & 1u);
  return r >> 16;
}

__device__ __forceinline__ float sigm(float x) {
  return 1.0f / (1.0f + __expf(-x));
}
__device__ __forceinline__ float tanh_fast(float x) {
  float e = __expf(-2.0f * fabsf(x));
  float t = (1.0f - e) / (1.0f + e);
  return copysignf(t, x);
}

// fused weight element: W_g = [Wv_g | Wc_g,c0 | Wc_g,c1], row n (=h out), col k
__device__ __forceinline__ float wget(const float* __restrict__ Wv,
                                      const float* __restrict__ Wc,
                                      unsigned g, unsigned n, unsigned k) {
  if (k < 256u)      return Wv[(g * 256u + n) * 256u + k];
  else if (k < 512u) return Wc[((g * 2u + 0u) * 256u + n) * 256u + (k - 256u)];
  else               return Wc[((g * 2u + 1u) * 256u + n) * 256u + (k - 512u)];
}

// ---------------------------------------------------------------------------
// Kernel 1: pack f32 weights into bf16 WMMA B-fragment order.
// Layout: wp[ ((g*24+kb)*16+nt)*32 + lane ][ 8 dwords ]
//   B matrix 32x16 (16-bit): lanes 0-15 -> col n=lane,   K = kb*32 + 0..15
//                            lanes 16-31-> col n=lane-16,K = kb*32 + 16..31
//   dword r holds K-pair (2r, 2r+1), low half = even K.
// ---------------------------------------------------------------------------
__global__ __launch_bounds__(256) void pack_weights_kernel(
    const float* __restrict__ Wv, const float* __restrict__ Wc,
    unsigned* __restrict__ wp) {
  unsigned id = blockIdx.x * 256u + threadIdx.x;     // 491520 total
  unsigned r    = id & 7u;
  unsigned lane = (id >> 3) & 31u;
  unsigned nt   = (id >> 8) & 15u;
  unsigned gkb  = id >> 12;                          // 0..119
  if (gkb >= 120u) return;
  unsigned g = gkb / 24u, kb = gkb % 24u;
  unsigned n = nt * 16u + (lane & 15u);
  unsigned k = kb * 32u + ((lane >> 4) << 4) + (r << 1);
  unsigned lo = f2bf(wget(Wv, Wc, g, n, k));
  unsigned hi = f2bf(wget(Wv, Wc, g, n, k + 1u));
  wp[id] = lo | (hi << 16);
}

// ---------------------------------------------------------------------------
// Per-gate GEMM over the staged LDS X-tile: s[4] = X_tile * W_g^T  (4 n-tiles)
// Single per-lane weight pointer, advanced by constant stride per kb; the 4
// n-tiles are reached via immediate offsets (whole packed array < 2^24 B).
// ---------------------------------------------------------------------------
__device__ __forceinline__ void gate_gemm(int g,
                                          const unsigned short* __restrict__ Xl,
                                          const uint4* __restrict__ wp,
                                          int mt, int ntb, int lane,
                                          v8f s[4]) {
  const int l15 = lane & 15;
  const int hsl = lane >> 4;                 // lane half selects K sub-block
  const int arow = mt * 16 + l15;            // A: M = lane (mod 16)
#pragma unroll
  for (int j = 0; j < 4; ++j) s[j] = (v8f){0.f,0.f,0.f,0.f,0.f,0.f,0.f,0.f};

  // base pointers (kb = 0); one 64-bit add each per kb iteration
  const unsigned short* ap = Xl + arow * XSTRIDE + hsl * 16;
  const uint4* bp = wp + ((size_t)((g * KB_N * NT_N + ntb) * 32 + lane) << 1);

  for (int kb = 0; kb < KB_N; ++kb) {
    // A fragment 16x32 bf16: q0 = K[hsl*16 .. +7], q1 = K[hsl*16+8 .. +7]
    union { v16bf v; uint4 q[2]; } A;
    A.q[0] = *(const uint4*)(ap);
    A.q[1] = *(const uint4*)(ap + 8);
#pragma unroll
    for (int j = 0; j < 4; ++j) {
      union { v16bf v; uint4 q[2]; } Bf;
      // n-tile stride = 32 lanes * 32B = 1024B -> immediate offsets
      Bf.q[0] = bp[j * 64];
      Bf.q[1] = bp[j * 64 + 1];
      s[j] = __builtin_amdgcn_wmma_f32_16x16x32_bf16(
          false, A.v, false, Bf.v, (short)0, s[j], false, false);
    }
    ap += 32;            // next 32 K values in the LDS row
    bp += NT_N * 64;     // next kb block: 16 n-tiles * 1KB = 16KB
  }
}

// ---------------------------------------------------------------------------
// Kernel 2: fused TreeLSTM cell. One block = 32 rows of B, all 256 H columns.
// Gate phases (f0, f1, m, i, o) reuse one LDS X-tile; forget-accumulator and
// tanh terms carried in VGPRs across phases.
// ---------------------------------------------------------------------------
__global__ __launch_bounds__(256) void treecell_kernel(
    const float* __restrict__ x,    // [B, 256]
    const float* __restrict__ hs,   // [2, B, 256]
    const float* __restrict__ cs,   // [2, B, 256]
    const float* __restrict__ bv,   // [5, 256]
    const uint4* __restrict__ wp,   // packed bf16 weights
    float* __restrict__ outh,       // [B, 256]
    float* __restrict__ outs) {     // [B, 256]
  __shared__ unsigned short Xl[M_TILE * XSTRIDE];

  const int  tid = threadIdx.x;
  const long rowbase = (long)blockIdx.x * M_TILE;

  // ---- Stage X = [input | h0 | h1] tile as bf16 into LDS ----
  const float4* srcs[3] = {
      (const float4*)(x + rowbase * H_SZ),
      (const float4*)(hs + rowbase * H_SZ),
      (const float4*)(hs + BH + rowbase * H_SZ)};
#pragma unroll
  for (int sidx = 0; sidx < 3; ++sidx) {
    const float4* src = srcs[sidx];
#pragma unroll
    for (int it = 0; it < 8; ++it) {
      int idx = it * 256 + tid;        // 0..2047 float4s
      int row = idx >> 6;              // 64 float4 per row
      int c4  = idx & 63;
      float4 f = src[row * 64 + c4];
      uint2 pk;
      pk.x = f2bf(f.x) | (f2bf(f.y) << 16);
      pk.y = f2bf(f.z) | (f2bf(f.w) << 16);
      *(uint2*)&Xl[row * XSTRIDE + sidx * 256 + c4 * 4] = pk;
    }
  }
  __syncthreads();

  const int wave = tid >> 5, lane = tid & 31;
  const int mt  = wave & 1;            // 2 m-tiles
  const int ntb = (wave >> 1) * 4;     // 4 n-tiles per wave
  const int l15 = lane & 15;
  const int hsel = lane >> 4;
  // C/D layout: dword r, lanes0-15 -> M=r,N=lane; lanes16-31 -> M=r+8,N=lane-16
  const long obase = (rowbase + mt * 16 + hsel * 8) * H_SZ + ntb * 16 + l15;

  v8f s[4];
  float accv[32];   // running forget sum, later new_state path
  float tmv[32];    // tanh(m), later tanh(new_state)

  // ---- phase: forget gate child 0 (g=3) ----
  gate_gemm(3, Xl, wp, mt, ntb, lane, s);
#pragma unroll
  for (int j = 0; j < 4; ++j) {
    float bb = bv[3 * 256 + (ntb + j) * 16 + l15];
#pragma unroll
    for (int r = 0; r < 8; ++r) {
      long idx = obase + (long)r * H_SZ + j * 16;
      accv[j * 8 + r] = sigm(s[j][r] + bb) * cs[idx];
    }
  }

  // ---- phase: forget gate child 1 (g=4) ----
  gate_gemm(4, Xl, wp, mt, ntb, lane, s);
#pragma unroll
  for (int j = 0; j < 4; ++j) {
    float bb = bv[4 * 256 + (ntb + j) * 16 + l15];
#pragma unroll
    for (int r = 0; r < 8; ++r) {
      long idx = obase + (long)r * H_SZ + j * 16;
      accv[j * 8 + r] += sigm(s[j][r] + bb) * cs[BH + idx];
    }
  }

  // ---- phase: memory gate m (g=2) ----
  gate_gemm(2, Xl, wp, mt, ntb, lane, s);
#pragma unroll
  for (int j = 0; j < 4; ++j) {
    float bb = bv[2 * 256 + (ntb + j) * 16 + l15];
#pragma unroll
    for (int r = 0; r < 8; ++r)
      tmv[j * 8 + r] = tanh_fast(s[j][r] + bb);
  }

  // ---- phase: input gate (g=0): new_state = sigm(i)*tanh(m) + forget ----
  gate_gemm(0, Xl, wp, mt, ntb, lane, s);
#pragma unroll
  for (int j = 0; j < 4; ++j) {
    float bb = bv[0 * 256 + (ntb + j) * 16 + l15];
#pragma unroll
    for (int r = 0; r < 8; ++r) {
      long idx = obase + (long)r * H_SZ + j * 16;
      float ns = sigm(s[j][r] + bb) * tmv[j * 8 + r] + accv[j * 8 + r];
      outs[idx] = ns;
      tmv[j * 8 + r] = tanh_fast(ns);
    }
  }

  // ---- phase: output gate (g=1): new_hidden = sigm(o)*tanh(new_state) ----
  gate_gemm(1, Xl, wp, mt, ntb, lane, s);
#pragma unroll
  for (int j = 0; j < 4; ++j) {
    float bb = bv[1 * 256 + (ntb + j) * 16 + l15];
#pragma unroll
    for (int r = 0; r < 8; ++r) {
      long idx = obase + (long)r * H_SZ + j * 16;
      outh[idx] = sigm(s[j][r] + bb) * tmv[j * 8 + r];
    }
  }
}

extern "C" void kernel_launch(void* const* d_in, const int* in_sizes, int n_in,
                              void* d_out, int out_size, void* d_ws, size_t ws_size,
                              hipStream_t stream) {
  const float* x  = (const float*)d_in[0];   // input        [B,256]
  const float* hs = (const float*)d_in[1];   // hidden_states[2,B,256]
  const float* cs = (const float*)d_in[2];   // cell_states  [2,B,256]
  const float* Wv = (const float*)d_in[3];   // [5,256,256]
  const float* bv = (const float*)d_in[4];   // [5,256]
  const float* Wc = (const float*)d_in[5];   // [5,2,256,256]

  float* outh = (float*)d_out;               // new_hidden
  float* outs = (float*)d_out + BH;          // new_state

  unsigned* wp = (unsigned*)d_ws;            // 1.97 MB packed bf16 weights

  // 5*24*16*32*8 = 491520 dwords -> 1920 blocks of 256
  hipLaunchKernelGGL(pack_weights_kernel, dim3(1920), dim3(256), 0, stream,
                     Wv, Wc, wp);
  hipLaunchKernelGGL(treecell_kernel, dim3(B_SZ / M_TILE), dim3(256), 0, stream,
                     x, hs, cs, bv, (const uint4*)d_ws, outh, outs);
}
